// ESSAttn_85435489452663
// MI455X (gfx1250) — compile-verified, matrix operand order checked by reading, not run on hardware
//
#include <hip/hip_runtime.h>
#include <hip/hip_bf16.h>
#include <math.h>

typedef __bf16 bf16_t;
typedef __attribute__((ext_vector_type(16))) __bf16 v16bf;
typedef __attribute__((ext_vector_type(8)))  __bf16 v8bf;
typedef __attribute__((ext_vector_type(8)))  float  v8f;

// 16-byte chunk type matching the async-LDS builtin's expected pointee
// ("int __attribute__((vector_size(16)))"), in global / shared address spaces.
typedef int b128i __attribute__((vector_size(16)));
typedef __attribute__((address_space(1))) b128i gb128i;
typedef __attribute__((address_space(3))) b128i lb128i;

#define EPS_SUM  1e-7f
#define EPS_NORM 1e-12f

// Problem dims (fixed by the reference)
#define BB 4
#define NN 4096
#define CC 1024

// CDNA5 async global->LDS staging (ASYNCcnt-tracked DMA path), if the
// toolchain exposes the builtin; otherwise fall back to VGPR round-trip.
#if __has_builtin(__builtin_amdgcn_global_load_async_to_lds_b128)
#define USE_ASYNC_LDS 1
#else
#define USE_ASYNC_LDS 0
#endif

__device__ __forceinline__ void wait_async0() {
#if __has_builtin(__builtin_amdgcn_s_wait_asynccnt)
    __builtin_amdgcn_s_wait_asynccnt(0);
#else
    asm volatile("s_wait_asynccnt 0" ::: "memory");
#endif
}

// ---------------------------------------------------------------------------
// fp32 -> bf16 convert
// ---------------------------------------------------------------------------
__global__ __launch_bounds__(256) void cvt_bf16_kernel(const float* __restrict__ in,
                                                       bf16_t* __restrict__ out, long n) {
    long i = (long)blockIdx.x * 256 + threadIdx.x;
    if (i < n) out[i] = (bf16_t)in[i];
}

// ---------------------------------------------------------------------------
// Tiled bf16 WMMA GEMM: D = A[M,K] * B[K,N] (+ epilogue), fp32 accumulate.
// Block tile 128x128x32, 256 threads = 8 wave32 (2x4), wave tile 64x32.
// A staged to LDS as [128][40] row-major (async DMA when available);
// B staged transposed to [128][40] ([n][k]) so both fragment loads are two
// ds_load_b128 per the CDNA5 16-bit A/B VGPR layout.
// ---------------------------------------------------------------------------
enum { EPI_BIAS = 0, EPI_ROWSCALE = 1, EPI_ADDV = 2 };

#define LDK 40  // 32 + 8 pad (16B-aligned rows, conflict-free frag loads)

template <int EPI, bool TRANSA>
__global__ __launch_bounds__(256) void gemm_bf16_kernel(
    const bf16_t* __restrict__ A, long strideA, int lda,
    const bf16_t* __restrict__ Bm, long strideB, int ldb,
    float* __restrict__ outF, bf16_t* __restrict__ outH, long strideO, int ldo,
    const float* __restrict__ aux, long strideAux, int ldaux,
    int M, int Nn, int K) {
    __shared__ bf16_t sA[2][128 * LDK];
    __shared__ bf16_t sB[2][128 * LDK];

    const int tid  = threadIdx.x;
    const int lane = tid & 31;
    const int wave = tid >> 5;
    const int wm   = wave >> 2;   // 0..1
    const int wn   = wave & 3;    // 0..3
    const int bm   = blockIdx.x * 128;
    const int bn   = blockIdx.y * 128;
    const int bz   = blockIdx.z;

    const bf16_t* Ab = A  + (long)bz * strideA;
    const bf16_t* Bb = Bm + (long)bz * strideB;

    v8f acc[4][2];
#pragma unroll
    for (int i = 0; i < 4; ++i)
#pragma unroll
        for (int j = 0; j < 2; ++j)
#pragma unroll
            for (int r = 0; r < 8; ++r) acc[i][j][r] = 0.0f;

    const int KT = K >> 5;

    auto stageA = [&](int buf, int kt) {
        if (!TRANSA) {
            // A row-major [M,K]
            const int row = tid >> 1;
            const int cb  = (tid & 1) << 4;
            const bf16_t* g = Ab + (long)(bm + row) * lda + kt * 32 + cb;
            bf16_t* l = &sA[buf][row * LDK + cb];
#if USE_ASYNC_LDS
            // memory -> LDS direct, no VGPR data (global_load_async_to_lds_b128)
            __builtin_amdgcn_global_load_async_to_lds_b128(
                (gb128i*)g, (lb128i*)l, 0, 0);
            __builtin_amdgcn_global_load_async_to_lds_b128(
                (gb128i*)(g + 8), (lb128i*)(l + 8), 0, 0);
#else
            v8bf x0 = *(const v8bf*)g;
            v8bf x1 = *(const v8bf*)(g + 8);
            *(v8bf*)l       = x0;
            *(v8bf*)(l + 8) = x1;
#endif
        } else {
            // A stored as [K,M] globally: transpose into LDS [m][k]
            const int k  = tid >> 3;
            const int rb = (tid & 7) << 4;
            const bf16_t* g = Ab + (long)(kt * 32 + k) * lda + bm + rb;
            v8bf x0 = *(const v8bf*)g;
            v8bf x1 = *(const v8bf*)(g + 8);
#pragma unroll
            for (int i = 0; i < 8; ++i) sA[buf][(rb + i) * LDK + k] = x0[i];
#pragma unroll
            for (int i = 0; i < 8; ++i) sA[buf][(rb + 8 + i) * LDK + k] = x1[i];
        }
    };
    auto stageB = [&](int buf, int kt) {
        // B row-major [K,N]: transpose into LDS [n][k]
        const int k  = tid >> 3;
        const int nb = (tid & 7) << 4;
        const bf16_t* g = Bb + (long)(kt * 32 + k) * ldb + bn + nb;
        v8bf x0 = *(const v8bf*)g;
        v8bf x1 = *(const v8bf*)(g + 8);
#pragma unroll
        for (int i = 0; i < 8; ++i) sB[buf][(nb + i) * LDK + k] = x0[i];
#pragma unroll
        for (int i = 0; i < 8; ++i) sB[buf][(nb + 8 + i) * LDK + k] = x1[i];
    };

    // Fragment load: lane L holds row (L&15), K = kb..kb+7 and kb+16..kb+23,
    // kb = (L>=16)*8  -> two 16B LDS loads.
    union VU { v16bf v; v8bf h[2]; };
    auto loadFrag = [&](const bf16_t* base, int row0) -> v16bf {
        const int r  = row0 + (lane & 15);
        const int kb = (lane >> 4) << 3;
        const bf16_t* p = base + r * LDK + kb;
        VU u;
        u.h[0] = *(const v8bf*)p;
        u.h[1] = *(const v8bf*)(p + 16);
        return u.v;
    };

    stageA(0, 0);
    stageB(0, 0);

    for (int kt = 0; kt < KT; ++kt) {
#if USE_ASYNC_LDS
        if (!TRANSA) wait_async0();  // staged buffer complete before barrier
#endif
        __syncthreads();
        const int cur = kt & 1;
        if (kt + 1 < KT) {
            stageA(cur ^ 1, kt + 1);
            stageB(cur ^ 1, kt + 1);
        }
        if (kt + 2 < KT) {  // global_prefetch_b8 for the tile after next
            __builtin_prefetch(Ab + (long)(TRANSA ? ((kt + 2) * 32) * lda + bm
                                                  : bm * (long)lda + (kt + 2) * 32), 0, 0);
            __builtin_prefetch(Bb + (long)((kt + 2) * 32) * ldb + bn, 0, 0);
        }

        v16bf af[4], bf[2];
#pragma unroll
        for (int i = 0; i < 4; ++i) af[i] = loadFrag(sA[cur], wm * 64 + i * 16);
#pragma unroll
        for (int j = 0; j < 2; ++j) bf[j] = loadFrag(sB[cur], wn * 32 + j * 16);

#pragma unroll
        for (int i = 0; i < 4; ++i)
#pragma unroll
            for (int j = 0; j < 2; ++j)
                acc[i][j] = __builtin_amdgcn_wmma_f32_16x16x32_bf16(
                    false, af[i], false, bf[j], (short)0, acc[i][j], false, false);
    }

    // Epilogue. C/D layout: lane L -> col n=(L&15); VGPR r -> row r+(L>=16)*8.
    const int nlo   = lane & 15;
    const int mhalf = (lane >> 4) << 3;
#pragma unroll
    for (int i = 0; i < 4; ++i) {
#pragma unroll
        for (int j = 0; j < 2; ++j) {
            const int gm0 = bm + wm * 64 + i * 16 + mhalf;
            const int gn  = bn + wn * 32 + j * 16 + nlo;
#pragma unroll
            for (int r = 0; r < 8; ++r) {
                const int gm = gm0 + r;
                float val = acc[i][j][r];
                if (EPI == EPI_BIAS) {
                    val += aux[gn];
                    outF[(long)bz * strideO + (long)gm * ldo + gn] = val;
                } else if (EPI == EPI_ROWSCALE) {
                    val *= aux[(long)bz * strideAux + gm];
                    outH[(long)bz * strideO + (long)gm * ldo + gn] = (bf16_t)val;
                } else {  // EPI_ADDV
                    val += aux[(long)bz * strideAux + (long)gm * ldaux + gn];
                    outH[(long)bz * strideO + (long)gm * ldo + gn] = (bf16_t)val;
                }
            }
        }
    }
}

// ---------------------------------------------------------------------------
// Per-token feature kernel: one block per (b,n). Reads qkv row (3*C fp32),
// produces q2 (row-sum + row-L2 normalized), k2' (row-sum normalized), v bf16.
// ---------------------------------------------------------------------------
__device__ __forceinline__ float blockSum(float v, float* sbuf, int tid) {
#pragma unroll
    for (int off = 16; off > 0; off >>= 1) v += __shfl_xor(v, off, 32);
    __syncthreads();  // protect sbuf from previous reduction's readers
    if ((tid & 31) == 0) sbuf[tid >> 5] = v;
    __syncthreads();
    float t = 0.0f;
#pragma unroll
    for (int i = 0; i < 8; ++i) t += sbuf[i];
    return t;
}

__global__ __launch_bounds__(256) void features_kernel(
    const float* __restrict__ qkv, bf16_t* __restrict__ q2o,
    bf16_t* __restrict__ k2o, bf16_t* __restrict__ vo) {
    __shared__ float sbuf[8];
    const int tid  = threadIdx.x;
    const long row = blockIdx.x;  // b*N + n
    const float* p = qkv + row * (3 * CC);

    float q[4], k[4], v[4];
#pragma unroll
    for (int j = 0; j < 4; ++j) {
        const int c = tid + j * 256;
        q[j] = p[c];
        k[j] = p[CC + c];
        v[j] = p[2 * CC + c];
    }

    const float qmean = blockSum(q[0] + q[1] + q[2] + q[3], sbuf, tid) * (1.0f / CC);
    const float kmean = blockSum(k[0] + k[1] + k[2] + k[3], sbuf, tid) * (1.0f / CC);

    float q2[4], k2[4];
    float qs = 0.0f, ks = 0.0f;
#pragma unroll
    for (int j = 0; j < 4; ++j) {
        const float qc = q[j] - qmean;
        const float kc = k[j] - kmean;
        q2[j] = qc * qc;
        k2[j] = kc * kc;
        qs += q2[j];
        ks += k2[j];
    }
    const float qinv = 1.0f / (blockSum(qs, sbuf, tid) + EPS_SUM);
    const float kinv = 1.0f / (blockSum(ks, sbuf, tid) + EPS_SUM);

    float t[4], tsq = 0.0f;
#pragma unroll
    for (int j = 0; j < 4; ++j) {
        t[j] = q2[j] * qinv;
        tsq += t[j] * t[j];
    }
    const float l2inv = 1.0f / fmaxf(sqrtf(blockSum(tsq, sbuf, tid)), EPS_NORM);

#pragma unroll
    for (int j = 0; j < 4; ++j) {
        const long c = tid + j * 256;
        q2o[row * CC + c] = (bf16_t)(t[j] * l2inv);
        k2o[row * CC + c] = (bf16_t)(k2[j] * kinv);
        vo[row * CC + c]  = (bf16_t)v[j];
    }
}

// ---------------------------------------------------------------------------
// Per-(b,c) column L2 norm of k2' over N tokens; folds t2's 1/sqrt(N) in.
// scale[b,c] = (1/max(||k2'[:,c]||, eps)) * (1/sqrt(N))
// ---------------------------------------------------------------------------
__global__ __launch_bounds__(256) void colscale_kernel(const bf16_t* __restrict__ k2p,
                                                       float* __restrict__ scale) {
    const int idx = blockIdx.x * 256 + threadIdx.x;  // b*C + c
    const int b = idx >> 10;
    const int c = idx & (CC - 1);
    const bf16_t* p = k2p + (long)b * NN * CC + c;
    float s = 0.0f;
    for (int n = 0; n < NN; ++n) {
        const float v = (float)p[(long)n * CC];
        s += v * v;
    }
    scale[idx] = (1.0f / fmaxf(sqrtf(s), EPS_NORM)) * 0.015625f;  // 1/sqrt(4096)
}

// ---------------------------------------------------------------------------
// Launch
// ---------------------------------------------------------------------------
extern "C" void kernel_launch(void* const* d_in, const int* in_sizes, int n_in,
                              void* d_out, int out_size, void* d_ws, size_t ws_size,
                              hipStream_t stream) {
    (void)in_sizes; (void)n_in; (void)out_size; (void)ws_size;
    const float* x    = (const float*)d_in[0];  // [4,4096,1024]
    const float* Wqkv = (const float*)d_in[1];  // [1024,3072]
    const float* bqkv = (const float*)d_in[2];  // [3072]
    const float* Wln  = (const float*)d_in[3];  // [1024,1024]
    const float* bln  = (const float*)d_in[4];  // [1024]
    float* out = (float*)d_out;                 // [4,4096,1024]

    // Workspace layout (bytes, 256B aligned)
    char* ws = (char*)d_ws;
    auto align_up = [](size_t v) { return (v + 255) & ~(size_t)255; };
    size_t off = 0;
    auto take = [&](size_t bytes) { char* p = ws + off; off += align_up(bytes); return p; };

    const long nX   = (long)BB * NN * CC;       // 16,777,216
    const long nQKV = (long)BB * NN * 3 * CC;   // 50,331,648
    const long nKV  = (long)BB * CC * CC;       // 4,194,304

    bf16_t* x16   = (bf16_t*)take(nX * 2);
    bf16_t* w1_16 = (bf16_t*)take((long)CC * 3 * CC * 2);
    bf16_t* wl_16 = (bf16_t*)take((long)CC * CC * 2);
    float*  qkv   = (float*) take(nQKV * 4);
    bf16_t* q2    = (bf16_t*)take(nX * 2);
    bf16_t* k2p   = (bf16_t*)take(nX * 2);
    bf16_t* v16   = (bf16_t*)take(nX * 2);
    bf16_t* kv16  = (bf16_t*)take(nKV * 2);
    bf16_t* u16   = (bf16_t*)take(nX * 2);
    float*  scale = (float*) take((long)BB * CC * 4);

    // 1) converts
    cvt_bf16_kernel<<<(unsigned)((nX + 255) / 256), 256, 0, stream>>>(x, x16, nX);
    cvt_bf16_kernel<<<(unsigned)(((long)CC * 3 * CC + 255) / 256), 256, 0, stream>>>(
        Wqkv, w1_16, (long)CC * 3 * CC);
    cvt_bf16_kernel<<<(unsigned)(((long)CC * CC + 255) / 256), 256, 0, stream>>>(
        Wln, wl_16, (long)CC * CC);

    // 2) qkv = x @ Wqkv + bqkv   [16384,3072] f32
    gemm_bf16_kernel<EPI_BIAS, false><<<dim3(128, 24, 1), 256, 0, stream>>>(
        x16, 0, CC, w1_16, 0, 3 * CC, qkv, nullptr, 0, 3 * CC,
        bqkv, 0, 0, BB * NN, 3 * CC, CC);

    // 3) per-token features
    features_kernel<<<BB * NN, 256, 0, stream>>>(qkv, q2, k2p, v16);

    // 4) column norms of k2' (+ fold 1/sqrt(N))
    colscale_kernel<<<(BB * CC) / 256, 256, 0, stream>>>(k2p, scale);

    // 5) kv = diag(scale) * (k2'^T @ v)   per batch [1024,1024] bf16
    gemm_bf16_kernel<EPI_ROWSCALE, true><<<dim3(8, 8, BB), 256, 0, stream>>>(
        k2p, (long)NN * CC, CC, v16, (long)NN * CC, CC,
        nullptr, kv16, (long)CC * CC, CC,
        scale, CC, 0, CC, CC, NN);

    // 6) u = v + q2 @ kv   per batch [4096,1024] bf16 (v read fp32 from qkv)
    gemm_bf16_kernel<EPI_ADDV, false><<<dim3(32, 8, BB), 256, 0, stream>>>(
        q2, (long)NN * CC, CC, kv16, (long)CC * CC, CC,
        nullptr, u16, (long)NN * CC, CC,
        qkv + 2 * CC, (long)NN * 3 * CC, 3 * CC, NN, CC, CC);

    // 7) out = u @ Wln + bln   [16384,1024] f32
    gemm_bf16_kernel<EPI_BIAS, false><<<dim3(128, 8, 1), 256, 0, stream>>>(
        u16, 0, CC, wl_16, 0, CC, out, nullptr, 0, CC,
        bln, 0, 0, BB * NN, CC, CC);
}
